// EmbedMatcher_12627203850793
// MI455X (gfx1250) — matrix operand order repaired and checked by reading, not compiled, not run on hardware
//
#include <hip/hip_runtime.h>
#include <hip/hip_bf16.h>
#include <math.h>

// ---------------------------------------------------------------------------
// Types for CDNA5 WMMA (wave32, 16x16x32 bf16 -> f32)
// ---------------------------------------------------------------------------
typedef __attribute__((ext_vector_type(16))) __bf16 v16bf;
typedef __attribute__((ext_vector_type(8)))  float  v8f;
typedef __attribute__((ext_vector_type(4)))  unsigned int u32x4;

union FragBF {
    v16bf v;
    u32x4 q[2];
};

static __device__ inline unsigned short f2bf(float x) {
    unsigned u = __builtin_bit_cast(unsigned, x);
    unsigned rnd = 0x7FFFu + ((u >> 16) & 1u);
    return (unsigned short)((u + rnd) >> 16);
}
static __device__ inline float sigf(float x) { return 1.0f / (1.0f + __expf(-x)); }

// Problem constants (match reference)
#define BQ   4096
#define SS   5
#define NBN  64
#define DD   128
#define DMM  256
#define HIDD 512
#define G4   2048   // 4*HID
// Padded GEMM row counts
#define M_GCN   8256   // >= 2*BQ + 2*SS, multiple of 64
#define M_SE    4160   // >= BQ + SS,     multiple of 64

// sched_group_barrier masks
#define SGB_WMMA 0x008
#define SGB_VMEM_RD 0x020

// ---------------------------------------------------------------------------
// Small utility kernels
// ---------------------------------------------------------------------------
__global__ void cvt_f32_bf16(const float* __restrict__ s, unsigned short* __restrict__ d, int n) {
    int i = blockIdx.x * blockDim.x + threadIdx.x;
    if (i < n) d[i] = f2bf(s[i]);
}
__global__ void zero_u16(unsigned short* p, int n) {
    int i = blockIdx.x * blockDim.x + threadIdx.x;
    if (i < n) p[i] = 0;
}
__global__ void zero_f32(float* p, int n) {
    int i = blockIdx.x * blockDim.x + threadIdx.x;
    if (i < n) p[i] = 0.0f;
}

// ---------------------------------------------------------------------------
// Gather-sum: xsum[row, 0:128] = sum_n emb[conn[row,n,0]];
//             xsum[row,128:256] = sum_n emb[conn[row,n,1]]   (stored bf16)
// One block (256 threads) per row; embedding table (102 MB) is L2-resident.
// ---------------------------------------------------------------------------
__global__ void gather_sum_kernel(const int* __restrict__ conn,
                                  const float* __restrict__ emb,
                                  unsigned short* __restrict__ xsum,
                                  int row_offset) {
    __shared__ int sconn[NBN * 2];
    const int row = blockIdx.x;
    const int t   = threadIdx.x;             // 0..255
    if (t < NBN * 2) sconn[t] = conn[row * NBN * 2 + t];
    __syncthreads();
    const int side = t >> 7;                 // 0: rel, 1: ent
    const int dim  = t & (DD - 1);
    float acc = 0.0f;
    #pragma unroll 4
    for (int n = 0; n < NBN; ++n) {
        int idx = sconn[n * 2 + side];
        acc += emb[(size_t)idx * DD + dim];
    }
    xsum[(size_t)(row_offset + row) * DMM + t] = f2bf(acc);
}

// ---------------------------------------------------------------------------
// WMMA strip, K double-buffered and pinned with sched_group_barrier so the
// scheduler keeps "loads of chunk k+1 in flight across WMMAs of chunk k".
// One wave accumulates a 16(M) x 64(N) tile of A(MxK) @ W^T; A row-major
// (lda=K), W (N x K) row-major (ldw=K).  K must be a multiple of 64.
// Fragments use contiguous k order: element s of lane half h holds k = 16h + s, same for A and W.
// ---------------------------------------------------------------------------
static __device__ inline void wmma_strip(const unsigned short* __restrict__ A, int lda,
                                         const unsigned short* __restrict__ W, int ldw,
                                         int m0, int n0, int K, v8f acc[4]) {
    const int lane = threadIdx.x & 31;
    const int half = lane >> 4;
    const int r    = lane & 15;
    const unsigned short* arow = A + (size_t)(m0 + r) * lda + 16 * half;
    const unsigned short* wrow = W + (size_t)(n0 + r) * ldw + 16 * half;

    FragBF a0, a1, b0[4], b1[4];
    auto loadA = [&](FragBF& f, int kk) {
        f.q[0] = *(const u32x4*)(arow + kk);
        f.q[1] = *(const u32x4*)(arow + kk + 8);
    };
    auto loadB = [&](FragBF* f, int kk) {
        #pragma unroll
        for (int t = 0; t < 4; ++t) {
            const unsigned short* wp = wrow + (size_t)t * 16 * ldw + kk;
            f[t].q[0] = *(const u32x4*)(wp);
            f[t].q[1] = *(const u32x4*)(wp + 8);
        }
    };

    loadA(a0, 0);
    loadB(b0, 0);
    for (int kk = 0; kk < K; kk += 64) {
        // issue next-chunk loads before consuming current fragments
        loadA(a1, kk + 32);
        loadB(b1, kk + 32);
        #pragma unroll
        for (int t = 0; t < 4; ++t)
            acc[t] = __builtin_amdgcn_wmma_f32_16x16x32_bf16(
                false, a0.v, false, b0[t].v, (short)0, acc[t], false, false);
        // group: 10 VMEM reads (a1/b1 chunk) then 4 WMMAs (a0/b0 chunk)
        __builtin_amdgcn_sched_group_barrier(SGB_VMEM_RD, 10, 0);
        __builtin_amdgcn_sched_group_barrier(SGB_WMMA,     4, 0);
        if (kk + 64 < K) {
            loadA(a0, kk + 64);
            loadB(b0, kk + 64);
        }
        #pragma unroll
        for (int t = 0; t < 4; ++t)
            acc[t] = __builtin_amdgcn_wmma_f32_16x16x32_bf16(
                false, a1.v, false, b1[t].v, (short)0, acc[t], false, false);
        __builtin_amdgcn_sched_group_barrier(SGB_VMEM_RD, 10, 0);
        __builtin_amdgcn_sched_group_barrier(SGB_WMMA,     4, 0);
    }
}

// Block = 128 threads = 4 waves tiled 2(M) x 2(N): block tile 32M x 128N.
// Halves duplicated A- and B-fragment traffic vs 4 waves sharing one n0.
#define WMMA_PROLOG(KVAL)                                            \
    const int wave = threadIdx.x >> 5;                               \
    const int lane = threadIdx.x & 31;                               \
    const int half = lane >> 4;                                      \
    const int r    = lane & 15;                                      \
    const int m0   = blockIdx.x * 32 + (wave >> 1) * 16;             \
    const int n0   = blockIdx.y * 128 + (wave & 1) * 64;             \
    v8f acc[4];                                                      \
    const v8f vz = {0,0,0,0,0,0,0,0};                                \
    acc[0]=vz; acc[1]=vz; acc[2]=vz; acc[3]=vz;                      \
    wmma_strip(A, lda, W, (KVAL), m0, n0, (KVAL), acc);

// GEMM1: GCN.  A = xsum (M_GCN x 256), W = gcn_w_bf (128 x 256).
// epilogue: tanh((acc + NB*bias[n]) / max(deg,1)) scattered into X (f32+bf16).
__global__ void gemm_gcn(const unsigned short* __restrict__ A, int lda,
                         const unsigned short* __restrict__ W,
                         const float* __restrict__ bias,
                         const int* __restrict__ qld, const int* __restrict__ qrd,
                         const int* __restrict__ sld, const int* __restrict__ srd,
                         float* __restrict__ Xf, unsigned short* __restrict__ Xb) {
    WMMA_PROLOG(DMM)
    #pragma unroll
    for (int t = 0; t < 4; ++t) {
        const int n = n0 + 16 * t + r;
        #pragma unroll
        for (int j = 0; j < 8; ++j) {
            const int m = m0 + j + 8 * half;
            int xrow, xcol, deg;
            if (m < BQ)                { xrow = m;                    xcol = n;       deg = qld[m]; }
            else if (m < 2*BQ)         { xrow = m - BQ;               xcol = 128 + n; deg = qrd[m - BQ]; }
            else if (m < 2*BQ + SS)    { xrow = BQ + (m - 2*BQ);      xcol = n;       deg = sld[m - 2*BQ]; }
            else if (m < 2*BQ + 2*SS)  { xrow = BQ + (m - 2*BQ - SS); xcol = 128 + n; deg = srd[m - 2*BQ - SS]; }
            else continue;
            float v = tanhf((acc[t][j] + (float)NBN * bias[n]) / fmaxf((float)deg, 1.0f));
            Xf[(size_t)xrow * DMM + xcol] = v;
            Xb[(size_t)xrow * DMM + xcol] = f2bf(v);
        }
    }
}

// GEMM2: proj1 + relu. A = X_bf (M_SE x 256), W = proj1_bf (512 x 256) -> h1_bf.
__global__ void gemm_relu(const unsigned short* __restrict__ A, int lda,
                          const unsigned short* __restrict__ W,
                          const float* __restrict__ bias,
                          unsigned short* __restrict__ Out, int ldo) {
    WMMA_PROLOG(DMM)
    #pragma unroll
    for (int t = 0; t < 4; ++t) {
        const int n = n0 + 16 * t + r;
        #pragma unroll
        for (int j = 0; j < 8; ++j) {
            const int m = m0 + j + 8 * half;
            Out[(size_t)m * ldo + n] = f2bf(fmaxf(acc[t][j] + bias[n], 0.0f));
        }
    }
}

// GEMM3: proj2 + residual. A = h1_bf (M_SE x 512), W = proj2_bf (256 x 512) -> z f32.
__global__ void gemm_residual(const unsigned short* __restrict__ A, int lda,
                              const unsigned short* __restrict__ W,
                              const float* __restrict__ bias,
                              const float* __restrict__ Res,
                              float* __restrict__ Z) {
    WMMA_PROLOG(HIDD)
    #pragma unroll
    for (int t = 0; t < 4; ++t) {
        const int n = n0 + 16 * t + r;
        #pragma unroll
        for (int j = 0; j < 8; ++j) {
            const int m = m0 + j + 8 * half;
            Z[(size_t)m * DMM + n] = acc[t][j] + bias[n] + Res[(size_t)m * DMM + n];
        }
    }
}

// GEMM4: xW = query_g @ w_ih^T + b_ih. A (BQ x 256), W (2048 x 256) -> f32.
__global__ void gemm_bias(const unsigned short* __restrict__ A, int lda,
                          const unsigned short* __restrict__ W,
                          const float* __restrict__ bias,
                          float* __restrict__ Out, int ldo) {
    WMMA_PROLOG(DMM)
    #pragma unroll
    for (int t = 0; t < 4; ++t) {
        const int n = n0 + 16 * t + r;
        #pragma unroll
        for (int j = 0; j < 8; ++j) {
            const int m = m0 + j + 8 * half;
            Out[(size_t)m * ldo + n] = acc[t][j] + bias[n];
        }
    }
}

// ---------------------------------------------------------------------------
// Fused LSTM step: gates GEMM (h_r @ w_hh^T) + bias + xW + cell nonlinearity.
// Block = 16 rows x 64 hidden columns; wave w computes gate w's 16x64 tile
// (global cols w*512 + j0..j0+63), stages it in LDS, then the block applies
// the cell update.  The 33.6 MB gates buffer never touches memory.
// h_r is ping-ponged (hr_in -> hr_out) to avoid cross-block RAW races.
// ---------------------------------------------------------------------------
__global__ void gemm_gates_cell(const unsigned short* __restrict__ hr_in,
                                const unsigned short* __restrict__ W,   // w_hh_bf (2048 x 512)
                                const float* __restrict__ bias,          // b_hh
                                const float* __restrict__ xW,            // (BQ x 2048)
                                float* __restrict__ c,                   // (BQ x 512)
                                const float* __restrict__ qg,            // (BQ x 256)
                                const unsigned short* __restrict__ sgb,  // (256) bf16
                                unsigned short* __restrict__ hr_out,     // (BQ x 512) bf16
                                float* __restrict__ h_out) {             // (BQ x 256)
    __shared__ float lg[4][16][64];
    const int wave = threadIdx.x >> 5;       // == gate index
    const int lane = threadIdx.x & 31;
    const int half = lane >> 4;
    const int r    = lane & 15;
    const int m0   = blockIdx.x * 16;
    const int jblk = blockIdx.y * 64;        // hidden-col block [0,512)
    const int n_base = wave * HIDD + jblk;   // column in 2048-wide gates

    v8f acc[4];
    const v8f vz = {0,0,0,0,0,0,0,0};
    acc[0]=vz; acc[1]=vz; acc[2]=vz; acc[3]=vz;
    wmma_strip(hr_in, HIDD, W, HIDD, m0, n_base, HIDD, acc);

    #pragma unroll
    for (int t = 0; t < 4; ++t) {
        const int nl = 16 * t + r;
        const int n  = n_base + nl;
        #pragma unroll
        for (int j = 0; j < 8; ++j) {
            const int ml = j + 8 * half;
            lg[wave][ml][nl] = acc[t][j] + bias[n] + xW[(size_t)(m0 + ml) * G4 + n];
        }
    }
    __syncthreads();

    // cell update: 128 threads x 8 elems = 16x64 tile
    #pragma unroll
    for (int e = 0; e < 8; ++e) {
        const int idx  = threadIdx.x * 8 + e;     // 0..1023
        const int rowl = idx >> 6;
        const int coll = idx & 63;
        const int b    = m0 + rowl;
        const int jcol = jblk + coll;             // 0..511
        const float gi = lg[0][rowl][coll];
        const float gf = lg[1][rowl][coll];
        const float gg = lg[2][rowl][coll];
        const float go = lg[3][rowl][coll];
        const size_t ci = (size_t)b * HIDD + jcol;
        float cc = sigf(gf) * c[ci] + sigf(gi) * tanhf(gg);
        c[ci] = cc;
        float hc = sigf(go) * tanhf(cc);
        if (jcol < DMM) {
            float h = qg[(size_t)b * DMM + jcol] + hc;
            h_out[(size_t)b * DMM + jcol] = h;
            hr_out[ci] = f2bf(h);
        } else {
            hr_out[ci] = sgb[jcol - DMM];
        }
    }
}

// ---------------------------------------------------------------------------
// LayerNorm (ddof=1) over z rows 0..BQ+SS-1 -> query_g (f32+bf16) / s_enc
// ---------------------------------------------------------------------------
__global__ void layernorm_kernel(const float* __restrict__ z,
                                 const float* __restrict__ g, const float* __restrict__ b,
                                 float* __restrict__ qgf, unsigned short* __restrict__ qgb,
                                 float* __restrict__ s_enc) {
    __shared__ float red[DMM];
    const int row = blockIdx.x;
    const int t   = threadIdx.x;
    float v = z[(size_t)row * DMM + t];
    red[t] = v; __syncthreads();
    for (int s = DMM / 2; s > 0; s >>= 1) { if (t < s) red[t] += red[t + s]; __syncthreads(); }
    float mu = red[0] / (float)DMM;
    __syncthreads();
    float d = v - mu;
    red[t] = d * d; __syncthreads();
    for (int s = DMM / 2; s > 0; s >>= 1) { if (t < s) red[t] += red[t + s]; __syncthreads(); }
    float sigma = sqrtf(red[0] / (float)(DMM - 1));
    float out = g[t] * d / (sigma + 1e-6f) + b[t];
    if (row < BQ) {
        qgf[(size_t)row * DMM + t] = out;
        qgb[(size_t)row * DMM + t] = f2bf(out);
    } else {
        s_enc[(size_t)(row - BQ) * DMM + t] = out;
    }
}

// support_g = mean over S rows; also ||support_g|| -> scal[0]
__global__ void support_mean_kernel(const float* __restrict__ s_enc,
                                    float* __restrict__ sgf, unsigned short* __restrict__ sgb,
                                    float* __restrict__ scal) {
    __shared__ float red[DMM];
    const int t = threadIdx.x;
    float acc = 0.0f;
    for (int s = 0; s < SS; ++s) acc += s_enc[(size_t)s * DMM + t];
    acc *= (1.0f / (float)SS);
    sgf[t] = acc;
    sgb[t] = f2bf(acc);
    red[t] = acc * acc; __syncthreads();
    for (int s = DMM / 2; s > 0; s >>= 1) { if (t < s) red[t] += red[t + s]; __syncthreads(); }
    if (t == 0) scal[0] = fmaxf(sqrtf(red[0]), 1e-12f);
}

// out[b] = (h[b] . sg) / (max(||h[b]||,1e-12) * ||sg||clamped)
__global__ void final_dot(const float* __restrict__ h, const float* __restrict__ sg,
                          const float* __restrict__ scal, float* __restrict__ out) {
    __shared__ float rd[DMM];
    __shared__ float rn[DMM];
    const int b = blockIdx.x;
    const int t = threadIdx.x;
    float hv = h[(size_t)b * DMM + t];
    rd[t] = hv * sg[t];
    rn[t] = hv * hv;
    __syncthreads();
    for (int s = DMM / 2; s > 0; s >>= 1) { if (t < s) { rd[t] += rd[t + s]; rn[t] += rn[t + s]; } __syncthreads(); }
    if (t == 0) out[b] = rd[0] / (fmaxf(sqrtf(rn[0]), 1e-12f) * scal[0]);
}

// ---------------------------------------------------------------------------
// Host launcher
// ---------------------------------------------------------------------------
extern "C" void kernel_launch(void* const* d_in, const int* in_sizes, int n_in,
                              void* d_out, int out_size, void* d_ws, size_t ws_size,
                              hipStream_t stream) {
    (void)in_sizes; (void)n_in; (void)out_size; (void)ws_size;

    const int*   q_l_conn = (const int*)  d_in[2];
    const int*   q_l_deg  = (const int*)  d_in[3];
    const int*   q_r_conn = (const int*)  d_in[4];
    const int*   q_r_deg  = (const int*)  d_in[5];
    const int*   s_l_conn = (const int*)  d_in[6];
    const int*   s_l_deg  = (const int*)  d_in[7];
    const int*   s_r_conn = (const int*)  d_in[8];
    const int*   s_r_deg  = (const int*)  d_in[9];
    const float* emb      = (const float*)d_in[10];
    const float* gcn_w    = (const float*)d_in[11];
    const float* gcn_b    = (const float*)d_in[12];
    const float* proj1_w  = (const float*)d_in[13];
    const float* proj1_b  = (const float*)d_in[14];
    const float* proj2_w  = (const float*)d_in[15];
    const float* proj2_b  = (const float*)d_in[16];
    const float* ln_g     = (const float*)d_in[17];
    const float* ln_b     = (const float*)d_in[18];
    const float* w_ih     = (const float*)d_in[19];
    const float* w_hh     = (const float*)d_in[20];
    const float* b_ih     = (const float*)d_in[21];
    const float* b_hh     = (const float*)d_in[22];
    float* out = (float*)d_out;

    // --- workspace layout ---
    size_t off = 0;
    auto alloc = [&](size_t bytes) -> char* {
        char* p = (char*)d_ws + off;
        off += (bytes + 255) & ~(size_t)255;
        return p;
    };
    unsigned short* w_gcn_bf = (unsigned short*)alloc((size_t)DD  * DMM  * 2);
    unsigned short* w_p1_bf  = (unsigned short*)alloc((size_t)HIDD* DMM  * 2);
    unsigned short* w_p2_bf  = (unsigned short*)alloc((size_t)DMM * HIDD * 2);
    unsigned short* w_ih_bf  = (unsigned short*)alloc((size_t)G4  * DMM  * 2);
    unsigned short* w_hh_bf  = (unsigned short*)alloc((size_t)G4  * HIDD * 2);
    unsigned short* xsum_bf  = (unsigned short*)alloc((size_t)M_GCN * DMM * 2);
    float*          X_f32    = (float*)         alloc((size_t)M_SE * DMM * 4);
    unsigned short* X_bf     = (unsigned short*)alloc((size_t)M_SE * DMM * 2);
    unsigned short* h1_bf    = (unsigned short*)alloc((size_t)M_SE * HIDD * 2);
    float*          z_f32    = (float*)         alloc((size_t)M_SE * DMM * 4);
    float*          qg_f32   = (float*)         alloc((size_t)BQ * DMM * 4);
    unsigned short* qg_bf    = (unsigned short*)alloc((size_t)BQ * DMM * 2);
    float*          s_enc    = (float*)         alloc((size_t)SS * DMM * 4);
    float*          sg_f32   = (float*)         alloc((size_t)DMM * 4);
    unsigned short* sg_bf    = (unsigned short*)alloc((size_t)DMM * 2);
    float*          scal     = (float*)         alloc(256);
    float*          xw_f32   = (float*)         alloc((size_t)BQ * G4 * 4);
    unsigned short* hr0      = (unsigned short*)alloc((size_t)BQ * HIDD * 2);
    unsigned short* hr1      = (unsigned short*)alloc((size_t)BQ * HIDD * 2);
    float*          c_f32    = (float*)         alloc((size_t)BQ * HIDD * 4);
    float*          h_f32    = (float*)         alloc((size_t)BQ * DMM * 4);

    const int T = 256;
    auto blks = [](int n, int t) { return (n + t - 1) / t; };

    // 1) weight conversion f32 -> bf16
    cvt_f32_bf16<<<blks(DD*DMM,   T), T, 0, stream>>>(gcn_w,   w_gcn_bf, DD*DMM);
    cvt_f32_bf16<<<blks(HIDD*DMM, T), T, 0, stream>>>(proj1_w, w_p1_bf,  HIDD*DMM);
    cvt_f32_bf16<<<blks(DMM*HIDD, T), T, 0, stream>>>(proj2_w, w_p2_bf,  DMM*HIDD);
    cvt_f32_bf16<<<blks(G4*DMM,   T), T, 0, stream>>>(w_ih,    w_ih_bf,  G4*DMM);
    cvt_f32_bf16<<<blks(G4*HIDD,  T), T, 0, stream>>>(w_hh,    w_hh_bf,  G4*HIDD);

    // 2) zero state + pad regions (fresh every launch: determinism)
    zero_u16<<<blks(BQ*HIDD, T), T, 0, stream>>>(hr0, BQ*HIDD);
    zero_f32<<<blks(BQ*HIDD, T), T, 0, stream>>>(c_f32, BQ*HIDD);
    {   // X rows [BQ+SS, M_SE) pad
        int padn = (M_SE - (BQ + SS)) * DMM;
        zero_f32<<<blks(padn, T), T, 0, stream>>>(X_f32 + (size_t)(BQ+SS)*DMM, padn);
        zero_u16<<<blks(padn, T), T, 0, stream>>>(X_bf  + (size_t)(BQ+SS)*DMM, padn);
    }
    {   // xsum rows [2BQ+2SS, M_GCN) pad
        int padn = (M_GCN - (2*BQ + 2*SS)) * DMM;
        zero_u16<<<blks(padn, T), T, 0, stream>>>(xsum_bf + (size_t)(2*BQ+2*SS)*DMM, padn);
    }

    // 3) gather-sum (rows: [0,B)=q_l, [B,2B)=q_r, [2B,2B+S)=s_l, [2B+S,2B+2S)=s_r)
    gather_sum_kernel<<<BQ, 256, 0, stream>>>(q_l_conn, emb, xsum_bf, 0);
    gather_sum_kernel<<<BQ, 256, 0, stream>>>(q_r_conn, emb, xsum_bf, BQ);
    gather_sum_kernel<<<SS, 256, 0, stream>>>(s_l_conn, emb, xsum_bf, 2*BQ);
    gather_sum_kernel<<<SS, 256, 0, stream>>>(s_r_conn, emb, xsum_bf, 2*BQ + SS);

    // 4) GCN GEMM + tanh epilogue -> X (f32 + bf16)   (block tile 32M x 128N)
    gemm_gcn<<<dim3(M_GCN/32, DD/128), 128, 0, stream>>>(
        xsum_bf, DMM, w_gcn_bf, gcn_b, q_l_deg, q_r_deg, s_l_deg, s_r_deg, X_f32, X_bf);

    // 5) proj1 + relu
    gemm_relu<<<dim3(M_SE/32, HIDD/128), 128, 0, stream>>>(X_bf, DMM, w_p1_bf, proj1_b, h1_bf, HIDD);

    // 6) proj2 + residual
    gemm_residual<<<dim3(M_SE/32, DMM/128), 128, 0, stream>>>(h1_bf, HIDD, w_p2_bf, proj2_b, X_f32, z_f32);

    // 7) layernorm over real rows
    layernorm_kernel<<<BQ + SS, DMM, 0, stream>>>(z_f32, ln_g, ln_b, qg_f32, qg_bf, s_enc);

    // 8) support mean + norm
    support_mean_kernel<<<1, DMM, 0, stream>>>(s_enc, sg_f32, sg_bf, scal);

    // 9) xW = query_g @ w_ih^T + b_ih
    gemm_bias<<<dim3(BQ/32, G4/128), 128, 0, stream>>>(qg_bf, DMM, w_ih_bf, b_ih, xw_f32, G4);

    // 10) 4 fused LSTM steps (gates GEMM + cell in one kernel, h_r ping-pong)
    for (int step = 0; step < 4; ++step) {
        const unsigned short* hin  = (step & 1) ? hr1 : hr0;
        unsigned short*       hout = (step & 1) ? hr0 : hr1;
        gemm_gates_cell<<<dim3(BQ/16, HIDD/64), 128, 0, stream>>>(
            hin, w_hh_bf, b_hh, xw_f32, c_f32, qg_f32, sg_bf, hout, h_f32);
    }

    // 11) cosine score
    final_dot<<<BQ, DMM, 0, stream>>>(h_f32, sg_f32, scal, out);
}
